// MultiHeadLinearAttention_33715493274040
// MI455X (gfx1250) — matrix operand order repaired and checked by reading, not compiled, not run on hardware
//
#include <hip/hip_runtime.h>
#include <hip/hip_bf16.h>

typedef __attribute__((ext_vector_type(16))) __bf16 v16bf;
typedef __attribute__((ext_vector_type(8)))  __bf16 v8bf;
typedef __attribute__((ext_vector_type(8)))  float  v8f;
typedef int v4i_vec __attribute__((vector_size(4 * sizeof(int))));

#define NSEQ 4096
#define DM   1024
#define NH   16
#define DK   64
#define LK   256

#if defined(__has_builtin)
#if __has_builtin(__builtin_amdgcn_global_load_async_to_lds_b128) && \
    __has_builtin(__builtin_amdgcn_s_wait_asynccnt)
#define USE_ASYNC 1
#endif
#endif

#ifdef USE_ASYNC
#define ASYNC_WAIT(n) __builtin_amdgcn_s_wait_asynccnt(n)
#else
#define ASYNC_WAIT(n) ((void)0)
#endif

// Copy 16 bytes (8 bf16) global -> LDS. Async (ASYNCcnt-tracked) when available.
__device__ __forceinline__ void cp16(__bf16* lds_dst, const __bf16* g_src) {
#ifdef USE_ASYNC
    __builtin_amdgcn_global_load_async_to_lds_b128(
        (__attribute__((address_space(1))) v4i_vec*)g_src,
        (__attribute__((address_space(3))) v4i_vec*)lds_dst, 0, 0);
#else
    *(v8bf*)lds_dst = *(const v8bf*)g_src;
#endif
}

__device__ __forceinline__ v8f wmma_bf16(v16bf a, v16bf b, v8f c) {
    return __builtin_amdgcn_wmma_f32_16x16x32_bf16(false, a, false, b, (short)0, c, false, false);
}

// A fragment (16x32 bf16): lane holds row m=lane&15; K runs {8g..8g+7},{16+8g..}.
__device__ __forceinline__ v16bf frag_a_lds(const __bf16* base, int ld, int lane) {
    int m = lane & 15, g = (lane >> 4) & 1;
    const __bf16* p = base + m * ld + g * 8;
    v16bf a;
    *(v8bf*)&a       = *(const v8bf*)(p);
    *((v8bf*)&a + 1) = *(const v8bf*)(p + 16);
    return a;
}

// B fragment (32x16 bf16): lane holds column c=lane&15; K runs {8g..8g+7},{16+8g..},
// in a column-major tile (stride ld per column).
__device__ __forceinline__ v16bf frag_b_lds(const __bf16* base, int ld, int c0, int lane) {
    int c = lane & 15, g = (lane >> 4) & 1;
    const __bf16* p = base + c * ld + c0 + g * 8;
    v16bf b;
    *(v8bf*)&b       = *(const v8bf*)(p);
    *((v8bf*)&b + 1) = *(const v8bf*)(p + 16);
    return b;
}

// A fragment straight from global f32 (row stride ld), converting to bf16.
__device__ __forceinline__ v16bf frag_a_gf32(const float* base, int ld, int lane) {
    int m = lane & 15, g = (lane >> 4) & 1;
    const float* p = base + m * ld + g * 8;
    v16bf a;
#pragma unroll
    for (int e = 0; e < 8; ++e) a[e] = (__bf16)p[e];
#pragma unroll
    for (int e = 0; e < 8; ++e) a[8 + e] = (__bf16)p[16 + e];
    return a;
}

// ---------------------------------------------------------------------------
// Kernel 0: WoT[c][k] = bf16(Wo[k][c]); 32x32 LDS tile transpose.
// ---------------------------------------------------------------------------
__global__ __launch_bounds__(256) void wot_kernel(const float* __restrict__ Wo,
                                                  __bf16* __restrict__ WoT) {
    __shared__ float t[32][33];
    int tx = threadIdx.x & 31, ty = threadIdx.x >> 5;   // 32 x 8
    int bx = blockIdx.x * 32, by = blockIdx.y * 32;
#pragma unroll
    for (int j = 0; j < 4; ++j)
        t[ty + 8 * j][tx] = Wo[(size_t)(by + ty + 8 * j) * DM + bx + tx];
    __syncthreads();
#pragma unroll
    for (int j = 0; j < 4; ++j)
        WoT[(size_t)(bx + ty + 8 * j) * DM + by + tx] = (__bf16)t[tx][ty + 8 * j];
}

// ---------------------------------------------------------------------------
// Kernel 1/2: projections, outputs stored TRANSPOSED so consumers copy
// contiguously:
//  MODE 0: KpT[bh][k=256][d=64] = (Kh^T We + be)^T
//  MODE 1: VpT[bh][d=64][k=256] = (Wf^T Vh + bf)^T
// ---------------------------------------------------------------------------
template<int MODE>
__global__ __launch_bounds__(256) void proj_kernel(const float* __restrict__ Xf,
                                                   const float* __restrict__ Yf,
                                                   const float* __restrict__ bias,
                                                   __bf16* __restrict__ out) {
    constexpr int M  = (MODE == 0) ? 64 : 256;
    constexpr int NC = (MODE == 0) ? 256 : 64;
    constexpr int NT = NC / 16;

    int bh = blockIdx.x, b = bh >> 4, h = bh & 15;
    const float* X; int ldx, xo;
    const float* Y; int ldy, yo;
    if (MODE == 0) { X = Xf + (size_t)b * NSEQ * DM; ldx = DM; xo = h * DK; Y = Yf; ldy = LK; yo = 0; }
    else           { X = Xf; ldx = LK; xo = 0; Y = Yf + (size_t)b * NSEQ * DM; ldy = DM; yo = h * DK; }
    __bf16* o = out + (size_t)bh * (M * NC);

    __shared__ __align__(16) __bf16 a_s[256 * 32];   // A tile, row-major [m][kk]
    __shared__ __align__(16) __bf16 b_s[256 * 32];   // B tile, col-major [c][kk]

    int tid = threadIdx.x, lane = tid & 31, w = tid >> 5, g = (lane >> 4) & 1;
    v8f acc[8];
#pragma unroll
    for (int i = 0; i < 8; ++i) acc[i] = v8f{0.f,0.f,0.f,0.f,0.f,0.f,0.f,0.f};

    for (int n0 = 0; n0 < NSEQ; n0 += 32) {
        for (int idx = tid; idx < M * 32; idx += 256) {   // coalesced over m
            int kk = idx / M, m = idx - kk * M;
            a_s[m * 32 + kk] = (__bf16)X[(size_t)(n0 + kk) * ldx + xo + m];
        }
        for (int idx = tid; idx < NC * 32; idx += 256) {  // coalesced over c
            int kk = idx / NC, c = idx - kk * NC;
            b_s[c * 32 + kk] = (__bf16)Y[(size_t)(n0 + kk) * ldy + yo + c];
        }
        __syncthreads();
#pragma unroll
        for (int i = 0; i < 8; ++i) {
            int t = w * 8 + i, rt = t / NT, ct = t - rt * NT;
            v16bf a  = frag_a_lds(a_s + rt * 16 * 32, 32, lane);
            v16bf bb = frag_b_lds(b_s + ct * 16 * 32, 32, 0, lane);
            acc[i] = wmma_bf16(a, bb, acc[i]);
        }
        __syncthreads();
    }
    // Transposed epilogue: element (row,col) -> o[col*M + row].
#pragma unroll
    for (int i = 0; i < 8; ++i) {
        int t = w * 8 + i, rt = t / NT, ct = t - rt * NT;
#pragma unroll
        for (int r = 0; r < 8; ++r) {
            int row = rt * 16 + r + 8 * g, col = ct * 16 + (lane & 15);
            float v = acc[i][r] + ((MODE == 0) ? bias[col] : bias[row]);
            o[(size_t)col * M + row] = (__bf16)v;
        }
    }
}

// ---------------------------------------------------------------------------
// Kernel 3: attention. grid = (N/128, H, B), 256 threads (8 waves).
// KpT/VpT staged with contiguous async b128 copies; wave w: 16 query rows.
// ---------------------------------------------------------------------------
__global__ __launch_bounds__(256) void attn_kernel(const float* __restrict__ Q,
                                                   const __bf16* __restrict__ KpT,
                                                   const __bf16* __restrict__ VpT,
                                                   __bf16* __restrict__ attn_out) {
    __shared__ __align__(16) __bf16 bufA[128 * 256]; // 64KB: KpT [256][64], later P [128][256]
    __shared__ __align__(16) __bf16 vp_s[64 * 256];  // 32KB: VpT [d][kk]

    int b = blockIdx.z, h = blockIdx.y, rb = blockIdx.x;
    int tid = threadIdx.x, lane = tid & 31, w = tid >> 5, g = (lane >> 4) & 1;
    const __bf16* kpT = KpT + (size_t)(b * NH + h) * LK * DK;
    const __bf16* vpT = VpT + (size_t)(b * NH + h) * DK * LK;

    for (int i = tid; i < (LK * DK) / 8; i += 256) cp16(bufA + i * 8, kpT + i * 8);
    for (int i = tid; i < (DK * LK) / 8; i += 256) cp16(vp_s + i * 8, vpT + i * 8);
    ASYNC_WAIT(0);
    __syncthreads();

    int row0 = rb * 128 + w * 16;
    const float* qbase = Q + ((size_t)b * NSEQ + row0) * DM + h * DK;

    v8f s[16];
#pragma unroll
    for (int i = 0; i < 16; ++i) s[i] = v8f{0.f,0.f,0.f,0.f,0.f,0.f,0.f,0.f};
#pragma unroll
    for (int kc = 0; kc < 2; ++kc) {
        v16bf a = frag_a_gf32(qbase + kc * 32, DM, lane);
#pragma unroll
        for (int ct = 0; ct < 16; ++ct) {
            v16bf bb = frag_b_lds(bufA + ct * 16 * DK, DK, kc * 32, lane);
            s[ct] = wmma_bf16(a, bb, s[ct]);
        }
    }
    // Softmax over k=256 per row; rows live in 16-lane halves (xor masks 1..8).
    const float scale = 0.125f; // 1/sqrt(64)
#pragma unroll
    for (int r = 0; r < 8; ++r) {
        float mx = -3.0e38f;
#pragma unroll
        for (int ct = 0; ct < 16; ++ct) mx = fmaxf(mx, s[ct][r]);
#pragma unroll
        for (int msk = 1; msk < 16; msk <<= 1) mx = fmaxf(mx, __shfl_xor(mx, msk, 32));
        float sum = 0.f;
#pragma unroll
        for (int ct = 0; ct < 16; ++ct) {
            float e = __expf(scale * (s[ct][r] - mx));
            s[ct][r] = e; sum += e;
        }
#pragma unroll
        for (int msk = 1; msk < 16; msk <<= 1) sum += __shfl_xor(sum, msk, 32);
        float inv = 1.f / sum;
#pragma unroll
        for (int ct = 0; ct < 16; ++ct) s[ct][r] *= inv;
    }
    __syncthreads();   // everyone done reading KpT from bufA
    __bf16* p = bufA + w * 16 * LK;   // wave-private P slab, A-layout row-major
#pragma unroll
    for (int ct = 0; ct < 16; ++ct)
#pragma unroll
        for (int r = 0; r < 8; ++r)
            p[(r + 8 * g) * LK + ct * 16 + (lane & 15)] = (__bf16)s[ct][r];
    __syncthreads();

    v8f o[4];
#pragma unroll
    for (int i = 0; i < 4; ++i) o[i] = v8f{0.f,0.f,0.f,0.f,0.f,0.f,0.f,0.f};
#pragma unroll
    for (int kc = 0; kc < 8; ++kc) {
        v16bf a = frag_a_lds(p + kc * 32, LK, lane);
#pragma unroll
        for (int ct = 0; ct < 4; ++ct) {
            v16bf bb = frag_b_lds(vp_s + ct * 16 * LK, LK, kc * 32, lane);
            o[ct] = wmma_bf16(a, bb, o[ct]);
        }
    }
    __bf16* outp = attn_out + ((size_t)b * NSEQ + row0) * DM + h * DK;
#pragma unroll
    for (int ct = 0; ct < 4; ++ct)
#pragma unroll
        for (int r = 0; r < 8; ++r)
            outp[(size_t)(r + 8 * g) * DM + ct * 16 + (lane & 15)] = (__bf16)o[ct][r];
}

// ---------------------------------------------------------------------------
// Kernel 4: out = Xa[16384,1024] x Wo + bo, both operands bf16 (Wo pre-
// transposed), 128x128 tiles, double-buffered async LDS staging, BK=32.
// ---------------------------------------------------------------------------
__global__ __launch_bounds__(256) void outgemm_kernel(const __bf16* __restrict__ Xa,
                                                      const __bf16* __restrict__ WoT,
                                                      const float* __restrict__ bo,
                                                      float* __restrict__ out) {
    __shared__ __align__(16) __bf16 a_s[2][128 * 32];  // [r][kk]        8KB x2
    __shared__ __align__(16) __bf16 w_s[2][128 * 32];  // col-major [c][kk]

    int tid = threadIdx.x, lane = tid & 31, w = tid >> 5, g = (lane >> 4) & 1;
    int rowbase = blockIdx.y * 128, colbase = blockIdx.x * 128;
    int wr = w & 3, wc = w >> 2;

    v8f acc[2][4];
#pragma unroll
    for (int i = 0; i < 2; ++i)
#pragma unroll
        for (int j = 0; j < 4; ++j) acc[i][j] = v8f{0.f,0.f,0.f,0.f,0.f,0.f,0.f,0.f};

    // Stage K-slice `it` into buffer `buf`: 4 async b128 copies per thread.
    auto stage = [&](int it, int buf) {
        int k0 = it * 32;
        const __bf16* xa = Xa  + (size_t)rowbase * DM + k0;
        const __bf16* wt = WoT + (size_t)colbase * DM + k0;
#pragma unroll
        for (int i = 0; i < 2; ++i) {
            int t2 = tid + i * 256;           // 512 transfers per operand
            int r = t2 >> 2, ch = t2 & 3;     // row, 16B chunk within 64B row
            cp16(&a_s[buf][r * 32 + ch * 8], xa + (size_t)r * DM + ch * 8);
        }
#pragma unroll
        for (int i = 0; i < 2; ++i) {
            int t2 = tid + i * 256;
            int c = t2 >> 2, ch = t2 & 3;
            cp16(&w_s[buf][c * 32 + ch * 8], wt + (size_t)c * DM + ch * 8);
        }
    };

    stage(0, 0);
    for (int it = 0; it < DM / 32; ++it) {
        int buf = it & 1;
        __syncthreads();                       // compute(it-1) reads finished
        if (it + 1 < DM / 32) { stage(it + 1, buf ^ 1); ASYNC_WAIT(4); }
        else                  { ASYNC_WAIT(0); }
        __syncthreads();                       // everyone's stage(it) landed
#pragma unroll
        for (int mt = 0; mt < 2; ++mt) {
            v16bf a = frag_a_lds(&a_s[buf][(wr * 32 + mt * 16) * 32], 32, lane);
#pragma unroll
            for (int nt = 0; nt < 4; ++nt) {
                v16bf bb = frag_b_lds(&w_s[buf][(wc * 64 + nt * 16) * 32], 32, 0, lane);
                acc[mt][nt] = wmma_bf16(a, bb, acc[mt][nt]);
            }
        }
    }
#pragma unroll
    for (int mt = 0; mt < 2; ++mt)
#pragma unroll
        for (int nt = 0; nt < 4; ++nt)
#pragma unroll
            for (int r = 0; r < 8; ++r) {
                int row = rowbase + wr * 32 + mt * 16 + r + 8 * g;
                int col = colbase + wc * 64 + nt * 16 + (lane & 15);
                out[(size_t)row * DM + col] = acc[mt][nt][r] + bo[col];
            }
}

extern "C" void kernel_launch(void* const* d_in, const int* in_sizes, int n_in,
                              void* d_out, int out_size, void* d_ws, size_t ws_size,
                              hipStream_t stream) {
    const float* K  = (const float*)d_in[0];
    const float* Q  = (const float*)d_in[1];
    const float* V  = (const float*)d_in[2];
    const float* We = (const float*)d_in[3];
    const float* be = (const float*)d_in[4];
    const float* Wf = (const float*)d_in[5];
    const float* bf = (const float*)d_in[6];
    const float* Wo = (const float*)d_in[7];
    const float* bo = (const float*)d_in[8];
    float* out = (float*)d_out;

    char* ws = (char*)d_ws;
    __bf16* kpT  = (__bf16*)ws;                      // 2 MB   [bh][k=256][d=64]
    __bf16* vpT  = (__bf16*)(ws + (2u << 20));       // 2 MB   [bh][d=64][k=256]
    __bf16* attn = (__bf16*)(ws + (4u << 20));       // 32 MB  [16384][1024]
    __bf16* woT  = (__bf16*)(ws + (36u << 20));      // 2 MB   [c=1024][k=1024]

    wot_kernel<<<dim3(DM / 32, DM / 32), 256, 0, stream>>>(Wo, woT);
    proj_kernel<0><<<dim3(64), 256, 0, stream>>>(K, We, be, kpT);
    proj_kernel<1><<<dim3(64), 256, 0, stream>>>(Wf, V, bf, vpT);
    attn_kernel<<<dim3(NSEQ / 128, NH, 4), 256, 0, stream>>>(Q, kpT, vpT, attn);
    outgemm_kernel<<<dim3(DM / 128, (4 * NSEQ) / 128), 256, 0, stream>>>(attn, woT, bo, out);
}
